// ShortestPathModel_72318659330649
// MI455X (gfx1250) — matrix operand order repaired and checked by reading, not compiled, not run on hardware
//
#include <hip/hip_runtime.h>
#include <hip/hip_bf16.h>
#include <stdint.h>

// ---------------------------------------------------------------------------
// MPNN (ShortestPathModel) for MI455X / gfx1250 (wave32, WMMA).
//
// d_in layout assumption (setup_inputs() insertion order, params flattened
// depth-first in insertion order):
//   0:x [N,1] f32   1:edge_index [2,E] i32   2:edge_attr [E,1] f32
//   3:lin_in.w [1,64]  4:lin_in.b [64]
//   per layer l (base 5+16*l):
//     +0 msg.lin1.w[129,64] +1 msg.lin1.b +2 msg.bn1.g +3 msg.bn1.b
//     +4 msg.lin2.w[64,64]  +5 msg.lin2.b +6 msg.bn2.g +7 msg.bn2.b
//     +8 upd.lin1.w[128,64] +9 upd.lin1.b +10 upd.bn1.g +11 upd.bn1.b
//     +12 upd.lin2.w[64,64] +13 upd.lin2.b +14 upd.bn2.g +15 upd.bn2.b
//   69:out_lin1.w 70:out_lin1.b 71:out_bn.g 72:out_bn.b 73:out_lin2.w 74:out_lin2.b
// ---------------------------------------------------------------------------

typedef __attribute__((ext_vector_type(16))) _Float16 v16h;
typedef __attribute__((ext_vector_type(8)))  _Float16 v8h;
typedef __attribute__((ext_vector_type(8)))  float    v8f;

#define EMB 64
#define WPB 8            // waves per block in GEMM kernels
#define BN_EPS 1e-5f
#define NMAT 17          // 4 layers * 4 weight mats + out_lin1
#define EGRID 1024       // persistent blocks for edge-sized GEMMs

__device__ __forceinline__ v8f wmma16(v16h a, v16h b, v8f c) {
  // D = A(16x32 f16) * B(32x16 f16) + C(16x16 f32)
  return __builtin_amdgcn_wmma_f32_16x16x32_f16(false, a, false, b, (short)0, c,
                                                false, false);
}

// Load 8 contiguous f32 and place as halfs a[O..O+7] (A-fragment chunk).
template <int O>
__device__ __forceinline__ void ld8f32(v16h& a, const float* p) {
  float4 u0 = ((const float4*)p)[0];
  float4 u1 = ((const float4*)p)[1];
  a[O + 0] = (_Float16)u0.x; a[O + 1] = (_Float16)u0.y;
  a[O + 2] = (_Float16)u0.z; a[O + 3] = (_Float16)u0.w;
  a[O + 4] = (_Float16)u1.x; a[O + 5] = (_Float16)u1.y;
  a[O + 6] = (_Float16)u1.z; a[O + 7] = (_Float16)u1.w;
}

// ---------------------------------------------------------------------------
// Weight -> WMMA B-fragment pre-pack.  Fragment f=(s,nt) covers K rows
// 32s..32s+31, cols 16nt..16nt+15.  Lane l holds N-col 16nt+l%16; half i holds
// K-row 32s+8*(l/16)+(i&7)+16*(i>>3) (one v16h = 32B per lane).
// ---------------------------------------------------------------------------
struct PrepDesc {
  const float* src[NMAT];
  _Float16*    dst[NMAT];
  int          fragbase[NMAT];
  int          totalfrags;
};

__global__ void prep_frags_kernel(PrepDesc d) {
  int t = blockIdx.x * blockDim.x + threadIdx.x;
  if (t >= d.totalfrags * 512) return;
  int f = t >> 9, e = t & 511;
  int lane = e >> 4, i = e & 15;
  int m = 0;
  #pragma unroll
  for (int q = 1; q < NMAT; ++q)
    if (f >= d.fragbase[q]) m = q;
  int fl = f - d.fragbase[m];
  int s = fl >> 2, nt = fl & 3;
  int k = 32 * s + 8 * (lane >> 4) + (i & 7) + 16 * (i >> 3);
  int n = 16 * nt + (lane & 15);
  d.dst[m][(size_t)fl * 512 + lane * 16 + i] = (_Float16)d.src[m][(size_t)k * EMB + n];
}

// ---------------------------------------------------------------------------
// Edge message lin1:  z[e,:] = W1d^T h[dst] + W1s^T h[src] + ea[e]*we + b1
// Persistent waves: each wave grid-strides over 16-edge tiles; the 16 B
// fragments (16 KB of weights) live in registers across all its tiles.
// BN column sum/sumsq accumulate in registers; reduced once at wave end.
// ---------------------------------------------------------------------------
__global__ __launch_bounds__(256)
void edge_lin1_kernel(const float* __restrict__ h, const int* __restrict__ ei,
                      const float* __restrict__ ea,
                      const _Float16* __restrict__ wfrag,
                      const float* __restrict__ w_raw,   // [129,64]; row 128 = edge w
                      const float* __restrict__ bias, _Float16* __restrict__ zout,
                      float* __restrict__ bnsum, int E) {
  __shared__ _Float16 zst[WPB][16 * EMB];
  __shared__ float csum[EMB], csq[EMB];
  const int tid = threadIdx.x;
  if (tid < EMB) { csum[tid] = 0.f; csq[tid] = 0.f; }
  __syncthreads();

  const int wave = tid >> 5, lane = tid & 31;
  const int lane15 = lane & 15, hk = lane >> 4;
  const int ntiles = E >> 4;

  // Register-resident B fragments + per-lane constants (reused across tiles).
  v16h bfr[16];
  #pragma unroll
  for (int f = 0; f < 16; ++f)
    bfr[f] = *(const v16h*)(wfrag + ((f * 32 + lane) << 4));
  float wl[4], bl[4];
  #pragma unroll
  for (int nt = 0; nt < 4; ++nt) {
    wl[nt] = w_raw[128 * EMB + 16 * nt + lane15];
    bl[nt] = bias[16 * nt + lane15];
  }
  float bnS[4] = {0.f, 0.f, 0.f, 0.f}, bnQ[4] = {0.f, 0.f, 0.f, 0.f};

  for (int tile = blockIdx.x * WPB + wave; tile < ntiles;
       tile += gridDim.x * WPB) {
    const int e0 = tile << 4;
    const int erow = e0 + lane15;                 // this lane's A-matrix row
    const int si = ei[erow];                      // src
    const int di = ei[E + erow];                  // dst
    const float* hd = h + (size_t)di * EMB;
    const float* hs = h + (size_t)si * EMB;
    v8f acc[4] = {};
    #pragma unroll
    for (int s = 0; s < 4; ++s) {                 // K step (32 per step)
      v16h a;
      const int kb1 = 32 * s + 8 * hk;            // halfs 0..7
      const int kb2 = kb1 + 16;                   // halfs 8..15
      ld8f32<0>(a, (kb1 < 64) ? (hd + kb1) : (hs + kb1 - 64));
      ld8f32<8>(a, (kb2 < 64) ? (hd + kb2) : (hs + kb2 - 64));
      #pragma unroll
      for (int nt = 0; nt < 4; ++nt)
        acc[nt] = wmma16(a, bfr[s * 4 + nt], acc[nt]);
    }
    // rank-1 edge-attr term + bias.  C elem: M = r + 8*hk, N = 16*nt + lane15
    float eav[8];
    #pragma unroll
    for (int r = 0; r < 8; ++r) eav[r] = ea[e0 + r + 8 * hk];
    #pragma unroll
    for (int nt = 0; nt < 4; ++nt) {
      #pragma unroll
      for (int r = 0; r < 8; ++r) {
        float v = acc[nt][r] + eav[r] * wl[nt] + bl[nt];
        bnS[nt] += v; bnQ[nt] += v * v;
        zst[wave][(r + 8 * hk) * EMB + 16 * nt + lane15] = (_Float16)v;
      }
    }
    // Wave-private staging region: same-wave DS ops are in-order, so the
    // coalesced readback needs no block barrier.
    const float4* sp = (const float4*)(&zst[wave][0]);
    float4* dp = (float4*)(zout + (size_t)e0 * EMB);
    #pragma unroll
    for (int i = 0; i < 4; ++i) dp[lane * 4 + i] = sp[lane * 4 + i];
  }

  // BN reduction: join M halves across lane pairs, then LDS, then global.
  #pragma unroll
  for (int nt = 0; nt < 4; ++nt) {
    float s1 = bnS[nt] + __shfl_xor(bnS[nt], 16, 32);
    float s2 = bnQ[nt] + __shfl_xor(bnQ[nt], 16, 32);
    if (hk == 0) {
      atomicAdd(&csum[16 * nt + lane15], s1);
      atomicAdd(&csq[16 * nt + lane15], s2);
    }
  }
  __syncthreads();
  if (tid < EMB) {
    atomicAdd(&bnsum[tid], csum[tid]);
    atomicAdd(&bnsum[EMB + tid], csq[tid]);
  }
}

// ---------------------------------------------------------------------------
// Row GEMM over 16-row tiles (persistent waves, register-resident B):
// out[r,:] = act(in[r,:]) @ W + bias, plus BN column sums.
// MODE 0: A = concat(in0_f32[row], in1_f32[row])       (KSTEPS=4, K=128)
// MODE 1: A = relu(a*in0_f16[row] + c)  (BN pre-act)   (KSTEPS=2, K=64, in-place OK)
// MODE 2: A = in0_f32[row]                             (KSTEPS=2, K=64)
// ---------------------------------------------------------------------------
template <int KSTEPS, int MODE>
__global__ __launch_bounds__(256)
void gemm_rows_kernel(const void* in0, const float* __restrict__ in1,
                      const float* __restrict__ bnac,
                      const _Float16* __restrict__ wfrag,
                      const float* __restrict__ bias, _Float16* out,
                      float* __restrict__ bnsum, int nrows) {
  __shared__ _Float16 zst[WPB][16 * EMB];
  __shared__ float csum[EMB], csq[EMB];
  __shared__ float sA[EMB], sC[EMB];
  const int tid = threadIdx.x;
  if (tid < EMB) {
    csum[tid] = 0.f; csq[tid] = 0.f;
    if (MODE == 1) { sA[tid] = bnac[tid]; sC[tid] = bnac[EMB + tid]; }
  }
  __syncthreads();

  const int wave = tid >> 5, lane = tid & 31;
  const int lane15 = lane & 15, hk = lane >> 4;
  const int ntiles = nrows >> 4;

  v16h bfr[KSTEPS * 4];
  #pragma unroll
  for (int f = 0; f < KSTEPS * 4; ++f)
    bfr[f] = *(const v16h*)(wfrag + ((f * 32 + lane) << 4));
  float bl[4];
  #pragma unroll
  for (int nt = 0; nt < 4; ++nt) bl[nt] = bias[16 * nt + lane15];
  float bnS[4] = {0.f, 0.f, 0.f, 0.f}, bnQ[4] = {0.f, 0.f, 0.f, 0.f};

  for (int tile = blockIdx.x * WPB + wave; tile < ntiles;
       tile += gridDim.x * WPB) {
    const int r0 = tile << 4;
    const int row = r0 + lane15;
    v8f acc[4] = {};
    #pragma unroll
    for (int s = 0; s < KSTEPS; ++s) {
      v16h a;
      const int kb1 = 32 * s + 8 * hk;
      const int kb2 = kb1 + 16;
      if constexpr (MODE == 1) {
        const _Float16* rp = (const _Float16*)in0 + (size_t)row * EMB;
        v8h lo = *(const v8h*)(rp + kb1);
        v8h hi = *(const v8h*)(rp + kb2);
        #pragma unroll
        for (int j = 0; j < 8; ++j) {
          a[j]     = (_Float16)fmaxf(fmaf(sA[kb1 + j], (float)lo[j], sC[kb1 + j]), 0.f);
          a[8 + j] = (_Float16)fmaxf(fmaf(sA[kb2 + j], (float)hi[j], sC[kb2 + j]), 0.f);
        }
      } else if constexpr (MODE == 0) {
        const float* h0 = (const float*)in0 + (size_t)row * EMB;
        const float* h1 = in1 + (size_t)row * EMB;
        ld8f32<0>(a, (kb1 < 64) ? (h0 + kb1) : (h1 + kb1 - 64));
        ld8f32<8>(a, (kb2 < 64) ? (h0 + kb2) : (h1 + kb2 - 64));
      } else {
        const float* h0 = (const float*)in0 + (size_t)row * EMB;
        ld8f32<0>(a, h0 + kb1);
        ld8f32<8>(a, h0 + kb2);
      }
      #pragma unroll
      for (int nt = 0; nt < 4; ++nt)
        acc[nt] = wmma16(a, bfr[s * 4 + nt], acc[nt]);
    }
    #pragma unroll
    for (int nt = 0; nt < 4; ++nt) {
      #pragma unroll
      for (int r = 0; r < 8; ++r) {
        float v = acc[nt][r] + bl[nt];
        bnS[nt] += v; bnQ[nt] += v * v;
        zst[wave][(r + 8 * hk) * EMB + 16 * nt + lane15] = (_Float16)v;
      }
    }
    const float4* sp = (const float4*)(&zst[wave][0]);
    float4* dp = (float4*)(out + (size_t)r0 * EMB);
    #pragma unroll
    for (int i = 0; i < 4; ++i) dp[lane * 4 + i] = sp[lane * 4 + i];
  }

  #pragma unroll
  for (int nt = 0; nt < 4; ++nt) {
    float s1 = bnS[nt] + __shfl_xor(bnS[nt], 16, 32);
    float s2 = bnQ[nt] + __shfl_xor(bnQ[nt], 16, 32);
    if (hk == 0) {
      atomicAdd(&csum[16 * nt + lane15], s1);
      atomicAdd(&csq[16 * nt + lane15], s2);
    }
  }
  __syncthreads();
  if (tid < EMB) {
    atomicAdd(&bnsum[tid], csum[tid]);
    atomicAdd(&bnsum[EMB + tid], csq[tid]);
  }
}

// ---------------------------------------------------------------------------
// BatchNorm finalize (training-mode biased stats): a=g*rsqrt(var+eps), c=b-a*mu.
// Re-zeros the sum buffer for the next accumulation.
// ---------------------------------------------------------------------------
__global__ void bn_finalize_kernel(float* bnsum, const float* __restrict__ g,
                                   const float* __restrict__ b, float invcount,
                                   float* bnac) {
  int n = threadIdx.x;  // 64 threads
  float s = bnsum[n], sq = bnsum[EMB + n];
  float mu = s * invcount;
  float var = fmaxf(sq * invcount - mu * mu, 0.f);
  float a = g[n] * rsqrtf(var + BN_EPS);
  bnac[n] = a;
  bnac[EMB + n] = b[n] - a * mu;
  bnsum[n] = 0.f;
  bnsum[EMB + n] = 0.f;
}

// segment_max of relu(a*m+c) onto aggr[dst].  Post-ReLU values are >= 0 and
// aggr is pre-zeroed, so integer atomicMax on the IEEE bit pattern is exact
// (and also reproduces the isneginf->0 rule for edge-less nodes).
__global__ void scatter_max_kernel(const _Float16* __restrict__ m,
                                   const float* __restrict__ bnac,
                                   const int* __restrict__ ei, float* aggr, int E) {
  int t = blockIdx.x * blockDim.x + threadIdx.x;
  if (t >= E * 16) return;
  int e = t >> 4;
  int cg = (t & 15) << 2;
  int d = ei[E + e];
  const _Float16* rp = m + (size_t)e * EMB + cg;
  int* ap = (int*)(aggr + (size_t)d * EMB + cg);
  #pragma unroll
  for (int j = 0; j < 4; ++j) {
    float v = fmaxf(fmaf(bnac[cg + j], (float)rp[j], bnac[EMB + cg + j]), 0.f);
    if (v > 0.f) atomicMax(ap + j, __float_as_int(v));
  }
}

__global__ void zero_kernel(float* p, int n) {
  int t = blockIdx.x * blockDim.x + threadIdx.x;
  if (t < n) p[t] = 0.f;
}

__global__ void lin_in_kernel(const float* __restrict__ x, const float* __restrict__ w,
                              const float* __restrict__ b, float* h, int N) {
  int t = blockIdx.x * blockDim.x + threadIdx.x;
  if (t >= N * EMB) return;
  int col = t & 63;
  h[t] = fmaf(x[t >> 6], w[col], b[col]);
}

__global__ void residual_kernel(float* h, const _Float16* __restrict__ un,
                                const float* __restrict__ bnac, int N) {
  int t = blockIdx.x * blockDim.x + threadIdx.x;
  if (t >= N * EMB) return;
  int col = t & 63;
  h[t] += fmaxf(fmaf(bnac[col], (float)un[t], bnac[EMB + col]), 0.f);
}

__global__ void out_final_kernel(const _Float16* __restrict__ u,
                                 const float* __restrict__ bnac,
                                 const float* __restrict__ w2,
                                 const float* __restrict__ b2, float* out, int N) {
  int i = blockIdx.x * blockDim.x + threadIdx.x;
  if (i >= N) return;
  const _Float16* rp = u + (size_t)i * EMB;
  float acc = 0.f;
  #pragma unroll 8
  for (int n = 0; n < EMB; ++n) {
    float o = fmaxf(fmaf(bnac[n], (float)rp[n], bnac[EMB + n]), 0.f);
    acc = fmaf(o, w2[n], acc);
  }
  out[i] = acc + b2[0];
}

// ---------------------------------------------------------------------------
extern "C" void kernel_launch(void* const* d_in, const int* in_sizes, int n_in,
                              void* d_out, int out_size, void* d_ws, size_t ws_size,
                              hipStream_t stream) {
  const int N = in_sizes[0];  // 100000 nodes
  const int E = in_sizes[2];  // 1250000 edges
  const float* x = (const float*)d_in[0];
  const int* ei = (const int*)d_in[1];
  const float* ea = (const float*)d_in[2];
  const float* lin_in_w = (const float*)d_in[3];
  const float* lin_in_b = (const float*)d_in[4];
  auto L = [&](int l, int off) { return (const float*)d_in[5 + 16 * l + off]; };
  const float* out1_w = (const float*)d_in[69];
  const float* out1_b = (const float*)d_in[70];
  const float* obn_g = (const float*)d_in[71];
  const float* obn_b = (const float*)d_in[72];
  const float* out2_w = (const float*)d_in[73];
  const float* out2_b = (const float*)d_in[74];
  (void)n_in; (void)ws_size;

  // Workspace carve-out (~225 MB total).
  char* ws = (char*)d_ws;
  auto carve = [&](size_t bytes) {
    char* p = ws;
    ws += (bytes + 255) & ~(size_t)255;
    return p;
  };
  float* h = (float*)carve((size_t)N * EMB * 4);
  float* aggr = (float*)carve((size_t)N * EMB * 4);
  _Float16* zm = (_Float16*)carve((size_t)E * EMB * 2);
  _Float16* un = (_Float16*)carve((size_t)N * EMB * 2);
  float* bnsum = (float*)carve(128 * 4);
  float* bnac = (float*)carve(128 * 4);
  _Float16* wfrag = (_Float16*)carve((size_t)(4 * 48 + 8) * 512 * 2);

  // Fragment prep descriptor: 17 matrices.
  PrepDesc pd;
  int fb = 0, mi = 0;
  auto addmat = [&](const float* src, int K) {
    pd.src[mi] = src;
    pd.fragbase[mi] = fb;
    pd.dst[mi] = wfrag + (size_t)fb * 512;
    fb += (K / 32) * 4;
    ++mi;
  };
  for (int l = 0; l < 4; ++l) {
    addmat(L(l, 0), 128);   // msg.lin1 (first 128 of 129 rows; edge row in VALU)
    addmat(L(l, 4), 64);    // msg.lin2
    addmat(L(l, 8), 128);   // upd.lin1
    addmat(L(l, 12), 64);   // upd.lin2
  }
  addmat(out1_w, 64);       // out_lin1
  pd.totalfrags = fb;       // 200 fragments

  const int ntE = E / 16, ntN = N / 16;
  const int gEfull = (ntE + WPB - 1) / WPB;
  const int gE = gEfull < EGRID ? gEfull : EGRID;   // persistent for edge GEMMs
  const int gN = (ntN + WPB - 1) / WPB;
  const int gNE = (N * EMB + 255) / 256;

  zero_kernel<<<1, 128, 0, stream>>>(bnsum, 128);
  prep_frags_kernel<<<(pd.totalfrags * 512 + 255) / 256, 256, 0, stream>>>(pd);
  lin_in_kernel<<<gNE, 256, 0, stream>>>(x, lin_in_w, lin_in_b, h, N);

  for (int l = 0; l < 4; ++l) {
    const _Float16* f_m1 = pd.dst[l * 4 + 0];
    const _Float16* f_m2 = pd.dst[l * 4 + 1];
    const _Float16* f_u1 = pd.dst[l * 4 + 2];
    const _Float16* f_u2 = pd.dst[l * 4 + 3];

    edge_lin1_kernel<<<gE, 256, 0, stream>>>(h, ei, ea, f_m1, L(l, 0), L(l, 1),
                                             zm, bnsum, E);
    bn_finalize_kernel<<<1, 64, 0, stream>>>(bnsum, L(l, 2), L(l, 3), 1.f / E, bnac);
    gemm_rows_kernel<2, 1><<<gE, 256, 0, stream>>>(zm, nullptr, bnac, f_m2, L(l, 5),
                                                   zm, bnsum, E);   // in-place safe
    bn_finalize_kernel<<<1, 64, 0, stream>>>(bnsum, L(l, 6), L(l, 7), 1.f / E, bnac);
    zero_kernel<<<gNE, 256, 0, stream>>>(aggr, N * EMB);
    scatter_max_kernel<<<(E * 16 + 255) / 256, 256, 0, stream>>>(zm, bnac, ei, aggr, E);
    gemm_rows_kernel<4, 0><<<gN, 256, 0, stream>>>(h, aggr, nullptr, f_u1, L(l, 9),
                                                   un, bnsum, N);
    bn_finalize_kernel<<<1, 64, 0, stream>>>(bnsum, L(l, 10), L(l, 11), 1.f / N, bnac);
    gemm_rows_kernel<2, 1><<<gN, 256, 0, stream>>>(un, nullptr, bnac, f_u2, L(l, 13),
                                                   un, bnsum, N);   // in-place safe
    bn_finalize_kernel<<<1, 64, 0, stream>>>(bnsum, L(l, 14), L(l, 15), 1.f / N, bnac);
    residual_kernel<<<gNE, 256, 0, stream>>>(h, un, bnac, N);
  }

  // Output head.
  gemm_rows_kernel<2, 2><<<gN, 256, 0, stream>>>(h, nullptr, nullptr, pd.dst[16],
                                                 out1_b, un, bnsum, N);
  bn_finalize_kernel<<<1, 64, 0, stream>>>(bnsum, obn_g, obn_b, 1.f / N, bnac);
  out_final_kernel<<<(N + 255) / 256, 256, 0, stream>>>(un, bnac, out2_w, out2_b,
                                                        (float*)d_out, N);
}
